// GraphAggregateLayer_81741817578210
// MI455X (gfx1250) — compile-verified, matrix operand order checked by reading, not run on hardware
//
#include <hip/hip_runtime.h>
#include <math.h>

// Problem constants (from reference): N=64, K=8, CH=32, H=W=64
#define NND   64
#define KNB   8
#define CHN   32
#define HH    64
#define WW    64
#define HWSZ  (HH * WW)          // 4096
#define XS_ELEMS (4 * 66 * 32)   // staged X tile: 4 rows x 66 padded cols x 32 ch = 8448
#define WS_ELEMS (9 * 32 * 32)   // staged W: 9 taps x 32co x 32ci = 9216
#define TPB   128                // 4 waves per block
#define BN_EPS 1e-5f

typedef __attribute__((ext_vector_type(16))) __bf16 v16bf;
typedef __attribute__((ext_vector_type(8)))  __bf16 v8bf;
typedef __attribute__((ext_vector_type(8)))  float  v8f;
typedef __attribute__((ext_vector_type(4)))  int    v4i;

#if defined(__has_builtin)
#if __has_builtin(__builtin_amdgcn_global_load_async_to_lds_b128) && \
    __has_builtin(__builtin_amdgcn_s_wait_asynccnt)
#define HAVE_ASYNC 1
#endif
#endif
#ifndef HAVE_ASYNC
#define HAVE_ASYNC 0
#endif

#define GAS __attribute__((address_space(1)))
#define LAS __attribute__((address_space(3)))

static __device__ __forceinline__ void async_copy_b128(const unsigned short* src,
                                                       unsigned short* dst) {
#if HAVE_ASYNC
    GAS v4i* g = (GAS v4i*)(uintptr_t)src;                 // flat == AS1 numerically
    LAS v4i* l = (LAS v4i*)(LAS void*)(void*)dst;          // generic -> AS3
    __builtin_amdgcn_global_load_async_to_lds_b128(g, l, 0, 0);
#else
    *(uint4*)dst = *(const uint4*)src;
#endif
}

static __device__ __forceinline__ void async_wait_all() {
#if HAVE_ASYNC
    __builtin_amdgcn_s_wait_asynccnt(0);
#endif
}

static __device__ __forceinline__ unsigned short f2bf(float f) {
    union { float f; unsigned u; } v; v.f = f;
    unsigned r = v.u + 0x7FFFu + ((v.u >> 16) & 1u);       // round-to-nearest-even
    return (unsigned short)(r >> 16);
}

// ---------------------------------------------------------------- utilities
__global__ __launch_bounds__(64) void zero_stats(float* p) {
    if (threadIdx.x < 64) p[threadIdx.x] = 0.0f;
}

// X (N,CH,H,W) f32 -> bf16 NHWC: Xbf[((n*64+h)*64+w)*32 + ci]
__global__ __launch_bounds__(256) void conv_x_nhwc(const float* __restrict__ X,
                                                   unsigned short* __restrict__ Xbf) {
    const int total = NND * CHN * HWSZ;     // 8,388,608
    for (int o = blockIdx.x * 256 + threadIdx.x; o < total; o += gridDim.x * 256) {
        int ci = o & 31;  int t  = o >> 5;
        int w  = t & 63;  int t2 = t >> 6;
        int h  = t2 & 63; int n  = t2 >> 6;
        Xbf[o] = f2bf(X[((size_t)(n * CHN + ci) * HH + h) * WW + w]);
    }
}

// Wagg (N,K,co,ci,3,3) f32 -> Wbf (N,K,tap,co,ci) bf16  (tap-major for WMMA A frags)
__global__ __launch_bounds__(256) void conv_w_bf16(const float* __restrict__ W,
                                                   unsigned short* __restrict__ Wbf) {
    const int total = NND * KNB * 9 * CHN * CHN;     // 4,718,592
    for (int o = blockIdx.x * 256 + threadIdx.x; o < total; o += gridDim.x * 256) {
        int ci = o & 31;  int o2 = o >> 5;
        int co = o2 & 31; int o3 = o2 >> 5;
        int t  = o3 % 9;  int e  = o3 / 9;           // e = node*8 + k
        int src = ((e * CHN + co) * CHN + ci) * 9 + t;
        Wbf[o] = f2bf(W[src]);
    }
}

// ------------------------------------------------- main edge-conv GEMM (WMMA)
// Block: node i x 128 spatial columns (2 image rows), 4 waves. Each wave owns a
// 32(co) x 32(col) output tile = four 16x16 f32 accumulators; the two A (weight)
// fragments are reused across both column tiles (2 LDS loads per WMMA). X tiles
// are double-buffered and staged via async-to-LDS b128 copies (ASYNCcnt path).
__global__ __launch_bounds__(TPB) void edge_conv_wmma(
    const unsigned short* __restrict__ Xbf,   // NHWC bf16
    const unsigned short* __restrict__ Wbf,   // [edge][tap][co][ci] bf16
    const float* __restrict__ bagg,
    const int* __restrict__ nn_idx,
    float* __restrict__ hidden) {

    __shared__ __align__(16) unsigned short Xs[2][XS_ELEMS];  // [row4][col66][ci32]
    __shared__ __align__(16) unsigned short Wsm[WS_ELEMS];    // [tap9][co32][ci32]
    __shared__ float bmean[CHN];

    const int tid  = threadIdx.x;
    const int wave = tid >> 5;             // 0..3
    const int lane = tid & 31;
    const int node = blockIdx.x >> 5;      // 64 nodes
    const int cb   = blockIdx.x & 31;      // column block (128 cols each)
    const int h0   = cb * 2;               // first image row of this block
    const int wrow = wave >> 1;            // 0/1: which of the 2 image rows
    const int wcol = (wave & 1) * 32;      // 32-col chunk within the row
    const int m    = lane & 15;
    const int g    = lane >> 4;            // lane group (K split per ISA layout)

    if (tid < CHN) {                       // bias mean over K (bagg is (N,K,CH))
        float s = 0.0f;
        for (int k = 0; k < KNB; ++k) s += bagg[(node * KNB + k) * CHN + tid];
        bmean[tid] = s * 0.125f;
    }

    // ---- pre-zero halo cells never written by staging (side cols, OOB rows)
    for (int e = tid; e < 256; e += TPB) {
        int row = e >> 6, rem = e & 63, side = rem >> 5, ci = rem & 31;
        int off = (row * 66 + side * 65) * CHN + ci;           // cols 0 and 65
        Xs[0][off] = 0; Xs[1][off] = 0;
    }
    if (h0 == 0)
        for (int e = tid; e < 66 * CHN; e += TPB) { Xs[0][e] = 0; Xs[1][e] = 0; }
    if (h0 == 62)
        for (int e = tid; e < 66 * CHN; e += TPB) {
            Xs[0][3 * 66 * CHN + e] = 0; Xs[1][3 * 66 * CHN + e] = 0;
        }

    int nbv[KNB];
    #pragma unroll
    for (int k = 0; k < KNB; ++k) nbv[k] = nn_idx[node * KNB + k];

    // stage interior of X tile (async b128: 8 channels of one pixel per chunk)
    auto stage_x = [&](int buf, int nb) {
        const unsigned short* Xn = Xbf + (size_t)nb * HWSZ * CHN;
        #pragma unroll
        for (int it = 0; it < 8; ++it) {                     // 1024 chunks / 128 thr
            int c   = it * TPB + tid;
            int row = c >> 8, rem = c & 255;
            int col = 1 + (rem >> 2), cig = rem & 3;
            int hh  = h0 - 1 + row;
            if ((unsigned)hh < (unsigned)HH) {
                const unsigned short* src = Xn + ((size_t)hh * WW + (col - 1)) * CHN + cig * 8;
                unsigned short* dst = &Xs[buf][(row * 66 + col) * CHN + cig * 8];
                async_copy_b128(src, dst);
            }
        }
    };
    // stage this edge's 9 tap matrices (contiguous 16B chunks)
    auto stage_w = [&](int k) {
        const unsigned short* wsrc = Wbf + (size_t)(node * KNB + k) * WS_ELEMS;
        #pragma unroll
        for (int it = 0; it < 9; ++it) {                     // 1152 chunks / 128 thr
            int e = it * TPB + tid;
            async_copy_b128(wsrc + e * 8, &Wsm[e * 8]);
        }
    };

    v8f acc00 = {};  // co  0..15, cols wcol+0..15
    v8f acc01 = {};  // co  0..15, cols wcol+16..31
    v8f acc10 = {};  // co 16..31, cols wcol+0..15
    v8f acc11 = {};  // co 16..31, cols wcol+16..31

    stage_x(0, nbv[0]);
    stage_w(0);
    async_wait_all();
    __syncthreads();

    for (int k = 0; k < KNB; ++k) {
        const int xb = k & 1;
        if (k + 1 < KNB) {
            stage_x(xb ^ 1, nbv[k + 1]);                       // overlap with WMMA
            __builtin_prefetch((const char*)(Wbf +
                (size_t)(node * KNB + k + 1) * WS_ELEMS) + tid * 144, 0, 3);
        }

        // ---- 9 taps: K = CH = 32 -> exactly one bf16 WMMA K-step per tap
        #pragma unroll
        for (int t = 0; t < 9; ++t) {
            const int r = t / 3, s = t - r * 3;
            // two B fragments (32ci x 16col each) for the 32-col tile
            const unsigned short* bp0 =
                &Xs[xb][((wrow + r) * 66 + (wcol + m + s)) * CHN + 8 * g];
            const unsigned short* bp1 = bp0 + 16 * CHN;        // +16 columns
            v8bf b0lo = *(const v8bf*)bp0, b0hi = *(const v8bf*)(bp0 + 16);
            v8bf b1lo = *(const v8bf*)bp1, b1hi = *(const v8bf*)(bp1 + 16);
            // two A fragments (16co x 32ci) shared by both column tiles
            const unsigned short* ap0 = &Wsm[(t * CHN + m) * CHN + 8 * g];
            const unsigned short* ap1 = &Wsm[(t * CHN + 16 + m) * CHN + 8 * g];
            v8bf a0lo = *(const v8bf*)ap0, a0hi = *(const v8bf*)(ap0 + 16);
            v8bf a1lo = *(const v8bf*)ap1, a1hi = *(const v8bf*)(ap1 + 16);
            v16bf bm0, bm1, am0, am1;
            #pragma unroll
            for (int e2 = 0; e2 < 8; ++e2) {
                bm0[e2] = b0lo[e2]; bm0[8 + e2] = b0hi[e2];
                bm1[e2] = b1lo[e2]; bm1[8 + e2] = b1hi[e2];
                am0[e2] = a0lo[e2]; am0[8 + e2] = a0hi[e2];
                am1[e2] = a1lo[e2]; am1[8 + e2] = a1hi[e2];
            }
            // pair WMMAs sharing the same A operand back-to-back
            acc00 = __builtin_amdgcn_wmma_f32_16x16x32_bf16(
                false, am0, false, bm0, (short)0, acc00, false, false);
            acc01 = __builtin_amdgcn_wmma_f32_16x16x32_bf16(
                false, am0, false, bm1, (short)0, acc01, false, false);
            acc10 = __builtin_amdgcn_wmma_f32_16x16x32_bf16(
                false, am1, false, bm0, (short)0, acc10, false, false);
            acc11 = __builtin_amdgcn_wmma_f32_16x16x32_bf16(
                false, am1, false, bm1, (short)0, acc11, false, false);
        }

        __syncthreads();                    // compute done -> Wsm reusable
        if (k + 1 < KNB) stage_w(k + 1);
        async_wait_all();                   // X(k+1) + W(k+1) landed in LDS
        __syncthreads();
    }

    // ---- epilogue: mean over K (x1/8), + bias mean, ELU, write hidden
    const int pbase = (h0 + wrow) * WW + wcol;
    float* outp = hidden + (size_t)node * CHN * HWSZ;
    #pragma unroll
    for (int v = 0; v < 8; ++v) {
        int row = v + 8 * g;                // C/D layout: VGPR v, lane group g
        int c0 = pbase + m;
        int c1 = c0 + 16;
        float h00 = acc00[v] * 0.125f + bmean[row];
        float h01 = acc01[v] * 0.125f + bmean[row];
        float h10 = acc10[v] * 0.125f + bmean[16 + row];
        float h11 = acc11[v] * 0.125f + bmean[16 + row];
        h00 = h00 > 0.0f ? h00 : expm1f(h00);
        h01 = h01 > 0.0f ? h01 : expm1f(h01);
        h10 = h10 > 0.0f ? h10 : expm1f(h10);
        h11 = h11 > 0.0f ? h11 : expm1f(h11);
        outp[(size_t)row * HWSZ + c0]        = h00;
        outp[(size_t)row * HWSZ + c1]        = h01;
        outp[(size_t)(16 + row) * HWSZ + c0] = h10;
        outp[(size_t)(16 + row) * HWSZ + c1] = h11;
    }
}

// ------------------------------------------------- BatchNorm statistics
__global__ __launch_bounds__(256) void bn_stats(const float* __restrict__ hidden,
                                                float* __restrict__ chsum,
                                                float* __restrict__ chsq) {
    __shared__ float ssum[256], ssq[256];
    const int c = blockIdx.x;        // channel
    const int slab = blockIdx.y;     // 0..7 (512 positions each)
    float s = 0.0f, q = 0.0f;
    for (int n = 0; n < NND; ++n) {
        const float* p = hidden + ((size_t)n * CHN + c) * HWSZ + slab * 512;
        for (int i = threadIdx.x; i < 512; i += 256) {
            float v = p[i]; s += v; q += v * v;
        }
    }
    ssum[threadIdx.x] = s; ssq[threadIdx.x] = q;
    __syncthreads();
    for (int off = 128; off > 0; off >>= 1) {
        if (threadIdx.x < off) {
            ssum[threadIdx.x] += ssum[threadIdx.x + off];
            ssq[threadIdx.x]  += ssq[threadIdx.x + off];
        }
        __syncthreads();
    }
    if (threadIdx.x == 0) {
        atomicAdd(&chsum[c], ssum[0]);
        atomicAdd(&chsq[c],  ssq[0]);
    }
}

// ------------------------------------------------- BatchNorm apply (in place)
__global__ __launch_bounds__(256) void bn_apply(float* __restrict__ out,
                                                const float* __restrict__ chsum,
                                                const float* __restrict__ chsq,
                                                const float* __restrict__ gamma,
                                                const float* __restrict__ beta) {
    __shared__ float scale[CHN], shift[CHN];
    if (threadIdx.x < CHN) {
        const int c = threadIdx.x;
        const float invc = 1.0f / ((float)NND * (float)HWSZ);
        float mu  = chsum[c] * invc;
        float var = chsq[c] * invc - mu * mu;
        float sc  = gamma[c] * rsqrtf(var + BN_EPS);
        scale[c] = sc;
        shift[c] = beta[c] - mu * sc;
    }
    __syncthreads();
    const size_t total = (size_t)NND * CHN * HWSZ;
    for (size_t i = (size_t)blockIdx.x * 256 + threadIdx.x; i < total;
         i += (size_t)gridDim.x * 256) {
        int c = (int)((i >> 12) & 31);   // HW = 4096 = 2^12
        out[i] = out[i] * scale[c] + shift[c];
    }
}

extern "C" void kernel_launch(void* const* d_in, const int* in_sizes, int n_in,
                              void* d_out, int out_size, void* d_ws, size_t ws_size,
                              hipStream_t stream) {
    const float* X     = (const float*)d_in[0];
    const float* Wagg  = (const float*)d_in[1];
    const float* bagg  = (const float*)d_in[2];
    const float* gamma = (const float*)d_in[3];
    const float* beta  = (const float*)d_in[4];
    const int*   nnidx = (const int*)d_in[5];
    float* out = (float*)d_out;

    // workspace: Xbf NHWC (16 MB) | Wbf (9 MB) | 64 floats of BN stats
    unsigned short* Xbf = (unsigned short*)d_ws;
    unsigned short* Wbf = Xbf + (size_t)NND * CHN * HWSZ;
    float* chsum = (float*)(Wbf + (size_t)NND * KNB * 9 * CHN * CHN);
    float* chsq  = chsum + CHN;

    hipLaunchKernelGGL(zero_stats, dim3(1), dim3(64), 0, stream, chsum);
    hipLaunchKernelGGL(conv_x_nhwc, dim3(4096), dim3(256), 0, stream, X, Xbf);
    hipLaunchKernelGGL(conv_w_bf16, dim3(4608), dim3(256), 0, stream, Wagg, Wbf);
    hipLaunchKernelGGL(edge_conv_wmma, dim3(2048), dim3(TPB), 0, stream,
                       Xbf, Wbf, bagg, nnidx, out);
    hipLaunchKernelGGL(bn_stats, dim3(32, 8), dim3(256), 0, stream, out, chsum, chsq);
    hipLaunchKernelGGL(bn_apply, dim3(2048), dim3(256), 0, stream,
                       out, chsum, chsq, gamma, beta);
}
